// NSM_17789754540887
// MI455X (gfx1250) — compile-verified, matrix operand order checked by reading, not run on hardware
//
#include <hip/hip_runtime.h>
#include <hip/hip_bf16.h>

// ---------------------------------------------------------------------------
// Sizes (fixed by the reference)
// ---------------------------------------------------------------------------
#define BG   50        // graphs
#define NPG  1000      // nodes per graph
#define NN   (BG*NPG)  // 50000 nodes
#define DEG  16
#define EE   (NN*DEG)  // 800000 edges
#define DD   128
#define PP   8
#define LL   24
#define VV   1024
#define OUTD 2000
#define NSTEP 4

typedef __bf16  bf16_t;
typedef bf16_t  v16bf __attribute__((ext_vector_type(16)));
typedef float   v8f   __attribute__((ext_vector_type(8)));

__device__ __forceinline__ float eluf(float x)  { return x > 0.0f ? x : (__expf(x) - 1.0f); }
__device__ __forceinline__ float sigmf(float x) { return 1.0f / (1.0f + __expf(-x)); }

// ---------------------------------------------------------------------------
// WMMA GEMM:  C[m,n] = sum_k A[m,k] * W[n,k]
//   A: M x 128, row stride lda (fp32)    W: 128x128 row-major [n][k] (fp32)
//   C: row stride ldc (fp32)
//   grid.y indexes a "p" slice: A += p*psA, W += p*psW, C += p*psC
// Per wave: one 16-row tile, 8 n-tiles x 4 k-blocks of v_wmma_f32_16x16x32_bf16.
// W is staged once per block into LDS in fragment-swizzled order so each lane
// reads its whole B fragment as one contiguous 32-byte LDS access.
// ---------------------------------------------------------------------------
__global__ __launch_bounds__(256) void nsm_gemm_wmma(
    const float* __restrict__ A, int lda,
    const float* __restrict__ W,
    float* __restrict__ C, int ldc,
    int Mtiles, int psA, int psW, int psC)
{
    const int p = blockIdx.y;
    A += (long)p * psA;
    W += (long)p * psW;
    C += (long)p * psC;

    __shared__ __align__(32) bf16_t Wl[8 * 4 * 32 * 16];   // [nt][kb][lane][16] = 32 KB

    const int tid = threadIdx.x;
    for (int idx = tid; idx < 8 * 4 * 32 * 16; idx += 256) {
        const int j    = idx & 15;
        const int lane = (idx >> 4) & 31;
        const int kb   = (idx >> 9) & 3;
        const int nt   = idx >> 11;
        const int h    = lane >> 4;
        const int n    = nt * 16 + (lane & 15);
        const int v    = j >> 1;
        const int k    = kb * 32 + ((v < 4) ? 0 : 16) + h * 8 + (v & 3) * 2 + (j & 1);
        Wl[idx] = (bf16_t)W[n * 128 + k];
    }
    __syncthreads();

    const int lane = tid & 31;
    const int wave = tid >> 5;
    const int m16  = lane & 15;   // A row within tile / C column within tile
    const int h    = lane >> 4;

    for (int tile = blockIdx.x * 8 + wave; tile < Mtiles; tile += gridDim.x * 8) {
        const long row0 = (long)tile * 16;

        // Load A fragments for the 4 k-blocks (fp32 -> bf16, per ISA A layout)
        v16bf a[4];
        const float* Arow = A + (row0 + m16) * (long)lda;
#pragma unroll
        for (int kb = 0; kb < 4; ++kb) {
#pragma unroll
            for (int v = 0; v < 8; ++v) {
                const int k = kb * 32 + ((v < 4) ? 0 : 16) + h * 8 + (v & 3) * 2;
                const float2 f2 = *(const float2*)(Arow + k);
                a[kb][2 * v]     = (bf16_t)f2.x;
                a[kb][2 * v + 1] = (bf16_t)f2.y;
            }
        }

#pragma unroll
        for (int nt = 0; nt < 8; ++nt) {
            v8f c = {};
#pragma unroll
            for (int kb = 0; kb < 4; ++kb) {
                const v16bf b = ((const v16bf*)Wl)[(nt * 4 + kb) * 32 + lane];
                c = __builtin_amdgcn_wmma_f32_16x16x32_bf16(
                        false, a[kb], false, b, (short)0, c, false, false);
            }
#pragma unroll
            for (int v = 0; v < 8; ++v) {
                const int m = v + 8 * h;                 // C row within tile
                C[(row0 + m) * (long)ldc + nt * 16 + m16] = c[v];
            }
        }
    }
}

// ---------------------------------------------------------------------------
// Concept tagger: per (b,l): softmax over V+1 concept logits, re-embed token.
// ---------------------------------------------------------------------------
__global__ __launch_bounds__(128) void nsm_tagger(
    const float* __restrict__ qtok, const float* __restrict__ vocab,
    const float* __restrict__ tdef, const float* __restrict__ tW,
    const int* __restrict__ qlen, float* __restrict__ tagged)
{
    const int bl = blockIdx.x;          // 0..B*L-1
    const int b  = bl / LL, l = bl % LL;
    const int d  = threadIdx.x;

    __shared__ float q[DD], qw[DD], logits[VV + 1], red[DD];

    q[d] = qtok[(long)(b * LL + l) * DD + d];
    __syncthreads();

    float acc = 0.0f;                                    // qw = q @ tagger_W
    for (int j = 0; j < DD; ++j) acc += q[j] * tW[j * DD + d];
    qw[d] = acc;
    __syncthreads();

    for (int v = d; v < VV; v += DD) {
        float s = 0.0f;
        const float* row = vocab + (long)v * DD;
        for (int e = 0; e < DD; ++e) s += qw[e] * row[e];
        logits[v] = s;
    }
    if (d == 0) {
        float s = 0.0f;
        for (int e = 0; e < DD; ++e) s += qw[e] * tdef[e];
        logits[VV] = s;
    }
    __syncthreads();

    float m = -1e30f;
    for (int v = d; v < VV + 1; v += DD) m = fmaxf(m, logits[v]);
    red[d] = m; __syncthreads();
    for (int s = 64; s > 0; s >>= 1) { if (d < s) red[d] = fmaxf(red[d], red[d + s]); __syncthreads(); }
    m = red[0]; __syncthreads();

    float zs = 0.0f;
    for (int v = d; v < VV + 1; v += DD) { float e = __expf(logits[v] - m); logits[v] = e; zs += e; }
    red[d] = zs; __syncthreads();
    for (int s = 64; s > 0; s >>= 1) { if (d < s) red[d] += red[d + s]; __syncthreads(); }
    const float inv = 1.0f / red[0];
    __syncthreads();

    float out = logits[VV] * inv * q[d];
    for (int v = 0; v < VV; ++v) out += logits[v] * inv * vocab[(long)v * DD + d];
    const float mask = (l < qlen[b]) ? 1.0f : 0.0f;
    tagged[(long)(b * LL + l) * DD + d] = out * mask;
}

// ---------------------------------------------------------------------------
// LSTM over L=24 steps; one block per batch element; thread d owns hidden d.
// ---------------------------------------------------------------------------
__global__ __launch_bounds__(128) void nsm_lstm(
    const float* __restrict__ tagged,
    const float* __restrict__ Wih, const float* __restrict__ Whh,
    const float* __restrict__ bih, const float* __restrict__ bhh,
    const int* __restrict__ qlen, float* __restrict__ encoded)
{
    const int b = blockIdx.x, d = threadIdx.x;
    __shared__ float x[DD], h[DD];
    float c = 0.0f, hd = 0.0f;
    h[d] = 0.0f;
    const int len = qlen[b];

    for (int t = 0; t < LL; ++t) {
        x[d] = tagged[(long)(b * LL + t) * DD + d];
        __syncthreads();
        float g[4];
#pragma unroll
        for (int gi = 0; gi < 4; ++gi) {
            const int row = gi * DD + d;
            float acc = bih[row] + bhh[row];
            const float* wi = Wih + (long)row * DD;
            const float* wh = Whh + (long)row * DD;
            for (int k = 0; k < DD; ++k) acc += wi[k] * x[k] + wh[k] * h[k];
            g[gi] = acc;
        }
        __syncthreads();
        if (t < len) {
            const float i = sigmf(g[0]), f = sigmf(g[1]);
            const float gg = tanhf(g[2]), o = sigmf(g[3]);
            c  = f * c + i * gg;
            hd = o * tanhf(c);
        }
        h[d] = hd;
        __syncthreads();
    }
    encoded[b * DD + d] = hd;
}

// ---------------------------------------------------------------------------
// RNN (4 steps) + attention over question tokens -> instructions (B,4,D)
// ---------------------------------------------------------------------------
__global__ __launch_bounds__(128) void nsm_rnn_attn(
    const float* __restrict__ encoded, const float* __restrict__ tagged,
    const float* __restrict__ Wih, const float* __restrict__ Whh,
    const float* __restrict__ bih, const float* __restrict__ bhh,
    const int* __restrict__ qlen, float* __restrict__ instr)
{
    const int b = blockIdx.x, d = threadIdx.x;
    __shared__ float enc[DD], hx[DD], att[LL];
    enc[d] = encoded[b * DD + d];
    hx[d]  = 0.0f;
    const int len = qlen[b];
    __syncthreads();

    for (int k = 0; k < NSTEP; ++k) {
        float acc = bih[d] + bhh[d];
        const float* wi = Wih + (long)d * DD;
        const float* wh = Whh + (long)d * DD;
        for (int j = 0; j < DD; ++j) acc += wi[j] * enc[j] + wh[j] * hx[j];
        __syncthreads();
        hx[d] = fmaxf(acc, 0.0f);
        __syncthreads();

        if (d < LL) {
            float s = 0.0f;
            const float* tl = tagged + (long)(b * LL + d) * DD;
            for (int j = 0; j < DD; ++j) s += hx[j] * tl[j];
            att[d] = (d < len) ? s : -1e30f;
        }
        __syncthreads();
        if (d == 0) {
            float m = -1e30f;
            for (int l = 0; l < LL; ++l) m = fmaxf(m, att[l]);
            float z = 0.0f;
            for (int l = 0; l < LL; ++l) { float e = __expf(att[l] - m); att[l] = e; z += e; }
            const float inv = 1.0f / z;
            for (int l = 0; l < LL; ++l) att[l] *= inv;
        }
        __syncthreads();
        float out = 0.0f;
        for (int l = 0; l < LL; ++l) out += att[l] * tagged[(long)(b * LL + l) * DD + d];
        instr[(long)(b * NSTEP + k) * DD + d] = out;
        __syncthreads();
    }
}

// ---------------------------------------------------------------------------
// psim = softmax(instr_t @ property_embeddings^T) over P+1
// ---------------------------------------------------------------------------
__global__ void nsm_psim(const float* __restrict__ instr, const float* __restrict__ pemb,
                         float* __restrict__ psim, int t)
{
    const int b = blockIdx.x, j = threadIdx.x;
    __shared__ float lg[PP + 1];
    if (j < PP + 1) {
        float s = 0.0f;
        const float* iv = instr + (long)(b * NSTEP + t) * DD;
        const float* pe = pemb + (long)j * DD;
        for (int e = 0; e < DD; ++e) s += iv[e] * pe[e];
        lg[j] = s;
    }
    __syncthreads();
    if (j == 0) {
        float m = lg[0];
        for (int i = 1; i < PP + 1; ++i) m = fmaxf(m, lg[i]);
        float z = 0.0f;
        for (int i = 0; i < PP + 1; ++i) { float e = __expf(lg[i] - m); lg[i] = e; z += e; }
        for (int i = 0; i < PP + 1; ++i) psim[b * (PP + 1) + i] = lg[i] / z;
    }
}

// ---------------------------------------------------------------------------
// Per-node: elu(instr_d * sum_p psim_p * prop_scored[n,p,d]) . w_node_score
// (node_scores never materialized)
// ---------------------------------------------------------------------------
__global__ __launch_bounds__(128) void nsm_node_score(
    const float* __restrict__ prop, const float* __restrict__ psim,
    const float* __restrict__ instr, const float* __restrict__ wns,
    const int* __restrict__ nidx, float* __restrict__ nscore, int t)
{
    const int n = blockIdx.x, d = threadIdx.x;
    const int b = nidx[n];
    float s = 0.0f;
#pragma unroll
    for (int p = 0; p < PP; ++p)
        s += psim[b * (PP + 1) + p] * prop[((long)n * PP + p) * DD + d];
    const float val  = eluf(instr[(long)(b * NSTEP + t) * DD + d] * s);
    const float part = val * wns[d];

    __shared__ float red[DD];
    red[d] = part; __syncthreads();
    for (int st = 64; st > 0; st >>= 1) { if (d < st) red[d] += red[d + st]; __syncthreads(); }
    if (d == 0) nscore[n] = red[0];
}

// ---------------------------------------------------------------------------
// Segment softmax over contiguous 1000-node graphs
// ---------------------------------------------------------------------------
__global__ __launch_bounds__(256) void nsm_seg_softmax(const float* __restrict__ scores,
                                                       float* __restrict__ out)
{
    const int b = blockIdx.x, tid = threadIdx.x;
    const float* s = scores + (long)b * NPG;
    __shared__ float red[256];

    float m = -1e30f;
    for (int i = tid; i < NPG; i += 256) m = fmaxf(m, s[i]);
    red[tid] = m; __syncthreads();
    for (int st = 128; st > 0; st >>= 1) { if (tid < st) red[tid] = fmaxf(red[tid], red[tid + st]); __syncthreads(); }
    m = red[0]; __syncthreads();

    float z = 0.0f;
    for (int i = tid; i < NPG; i += 256) z += __expf(s[i] - m);
    red[tid] = z; __syncthreads();
    for (int st = 128; st > 0; st >>= 1) { if (tid < st) red[tid] += red[tid + st]; __syncthreads(); }
    const float inv = 1.0f / red[0];
    for (int i = tid; i < NPG; i += 256) out[(long)b * NPG + i] = __expf(s[i] - m) * inv;
}

__global__ void nsm_zero(float* __restrict__ p, long n)
{
    const long i = (long)blockIdx.x * blockDim.x + threadIdx.x;
    if (i < n) p[i] = 0.0f;
}

// ---------------------------------------------------------------------------
// Edge pass: agg[dst] += dist[src] * elu(instr * edge_proj[e])  (f32 atomics)
// ---------------------------------------------------------------------------
__global__ __launch_bounds__(256) void nsm_edge_scatter(
    const float* __restrict__ eproj, const float* __restrict__ instr,
    const float* __restrict__ dist,
    const int* __restrict__ esrc, const int* __restrict__ edst,
    const int* __restrict__ ebi, float* __restrict__ agg, int t)
{
    const long e = (long)blockIdx.x * 2 + (threadIdx.x >> 7);
    const int  d = threadIdx.x & 127;
    const int  b = ebi[e];
    const float iv = instr[(long)(b * NSTEP + t) * DD + d];
    const float ep = eproj[e * DD + d];
    const float es = eluf(iv * ep);
    const float w  = dist[esrc[e]];
    atomicAdd(&agg[(long)edst[e] * DD + d], w * es);
}

__global__ __launch_bounds__(128) void nsm_rel_score(
    const float* __restrict__ agg, const float* __restrict__ wrs, float* __restrict__ rscore)
{
    const int n = blockIdx.x, d = threadIdx.x;
    __shared__ float red[DD];
    red[d] = agg[(long)n * DD + d] * wrs[d];
    __syncthreads();
    for (int st = 64; st > 0; st >>= 1) { if (d < st) red[d] += red[d + st]; __syncthreads(); }
    if (d == 0) rscore[n] = red[0];
}

__global__ void nsm_dist_init(const float* __restrict__ npg, const int* __restrict__ nidx,
                              float* __restrict__ dist)
{
    const int n = blockIdx.x * 256 + threadIdx.x;
    if (n < NN) dist[n] = 1.0f / npg[nidx[n]];
}

__global__ void nsm_dist_update(const float* __restrict__ psim, const float* __restrict__ prel,
                                const float* __restrict__ pst, const int* __restrict__ nidx,
                                float* __restrict__ dist)
{
    const int n = blockIdx.x * 256 + threadIdx.x;
    if (n < NN) {
        const int b = nidx[n];
        const float r = psim[b * (PP + 1) + PP];
        dist[n] = r * prel[n] + (1.0f - r) * pst[n];
    }
}

// ---------------------------------------------------------------------------
// aggregated[b,d] = sum_n dist[n] * sum_p sim_g[b,p] * node_attrs[n,p,d]
// 10 chunk-blocks per graph, register accumulate, one atomic per (block,d)
// ---------------------------------------------------------------------------
__global__ __launch_bounds__(128) void nsm_final_agg(
    const float* __restrict__ nattr, const float* __restrict__ psim,
    const float* __restrict__ dist, float* __restrict__ aggregated)
{
    const int blk = blockIdx.x;            // 0..499
    const int b = blk / 10, chunk = blk % 10;
    const int d = threadIdx.x;
    float sg[PP];
#pragma unroll
    for (int p = 0; p < PP; ++p) sg[p] = psim[b * (PP + 1) + p];

    float acc = 0.0f;
    const int n0 = b * NPG + chunk * 100;
    for (int i = 0; i < 100; ++i) {
        const long n = n0 + i;
        float s = 0.0f;
#pragma unroll
        for (int p = 0; p < PP; ++p) s += sg[p] * nattr[(n * PP + p) * DD + d];
        acc += dist[n] * s;
    }
    atomicAdd(&aggregated[b * DD + d], acc);
}

__global__ __launch_bounds__(256) void nsm_final_linear(
    const float* __restrict__ encoded, const float* __restrict__ aggregated,
    const float* __restrict__ linW, const float* __restrict__ linb, float* __restrict__ out)
{
    const int b = blockIdx.x, tid = threadIdx.x;
    __shared__ float x[2 * DD];
    if (tid < DD) x[tid] = encoded[b * DD + tid];
    else          x[tid] = aggregated[b * DD + (tid - DD)];
    __syncthreads();
    for (int o = tid; o < OUTD; o += 256) {
        float acc = linb[o];
        const float* w = linW + (long)o * (2 * DD);
        for (int j = 0; j < 2 * DD; ++j) acc += w[j] * x[j];
        out[(long)b * OUTD + o] = acc;
    }
}

// ---------------------------------------------------------------------------
extern "C" void kernel_launch(void* const* d_in, const int* in_sizes, int n_in,
                              void* d_out, int out_size, void* d_ws, size_t ws_size,
                              hipStream_t stream)
{
    (void)in_sizes; (void)n_in; (void)out_size; (void)ws_size;

    const float* node_attrs   = (const float*)d_in[0];
    const float* edge_attrs   = (const float*)d_in[1];
    const float* qtok         = (const float*)d_in[2];
    const float* vocab        = (const float*)d_in[3];
    const float* pemb         = (const float*)d_in[4];
    const float* npg          = (const float*)d_in[5];
    const float* tdef         = (const float*)d_in[6];
    const float* tW           = (const float*)d_in[7];
    const float* lstm_Wih     = (const float*)d_in[8];
    const float* lstm_Whh     = (const float*)d_in[9];
    const float* lstm_bih     = (const float*)d_in[10];
    const float* lstm_bhh     = (const float*)d_in[11];
    const float* rnn_Wih      = (const float*)d_in[12];
    const float* rnn_Whh      = (const float*)d_in[13];
    const float* rnn_bih      = (const float*)d_in[14];
    const float* rnn_bhh      = (const float*)d_in[15];
    const float* W_node_props = (const float*)d_in[16];
    const float* W_edge       = (const float*)d_in[17];
    const float* w_node_score = (const float*)d_in[18];
    const float* w_rel_score  = (const float*)d_in[19];
    const float* lin_W        = (const float*)d_in[20];
    const float* lin_b        = (const float*)d_in[21];
    const int*   node_indices = (const int*)d_in[22];
    const int*   edge_src     = (const int*)d_in[23];
    const int*   edge_dst     = (const int*)d_in[24];
    const int*   edge_bidx    = (const int*)d_in[25];
    const int*   qlen         = (const int*)d_in[26];

    char* wsb = (char*)d_ws;
    size_t off = 0;
    auto alloc = [&](size_t nfloats) -> float* {
        float* p = (float*)(wsb + off);
        off += ((nfloats * sizeof(float) + 255) / 256) * 256;
        return p;
    };
    float* prop       = alloc((size_t)NN * PP * DD);   // 204.8 MB
    float* eproj      = alloc((size_t)EE * DD);        // 409.6 MB
    float* agg        = alloc((size_t)NN * DD);        // 25.6 MB
    float* nscore     = alloc(NN);
    float* rscore     = alloc(NN);
    float* pstates    = alloc(NN);
    float* prel       = alloc(NN);
    float* dist       = alloc(NN);
    float* tagged     = alloc((size_t)BG * LL * DD);
    float* encoded    = alloc((size_t)BG * DD);
    float* instr      = alloc((size_t)BG * NSTEP * DD);
    float* psim       = alloc((size_t)BG * (PP + 1));
    float* aggregated = alloc((size_t)BG * DD);
    float* outp       = (float*)d_out;

    // -------- Phase A: question processing --------
    nsm_tagger  <<<dim3(BG * LL), dim3(128), 0, stream>>>(qtok, vocab, tdef, tW, qlen, tagged);
    nsm_lstm    <<<dim3(BG),      dim3(128), 0, stream>>>(tagged, lstm_Wih, lstm_Whh, lstm_bih, lstm_bhh, qlen, encoded);
    nsm_rnn_attn<<<dim3(BG),      dim3(128), 0, stream>>>(encoded, tagged, rnn_Wih, rnn_Whh, rnn_bih, rnn_bhh, qlen, instr);

    // -------- Big GEMMs on WMMA (bf16 in / f32 accumulate) --------
    nsm_gemm_wmma<<<dim3(2048, 1), dim3(256), 0, stream>>>(
        edge_attrs, DD, W_edge, eproj, DD, EE / 16, 0, 0, 0);
    nsm_gemm_wmma<<<dim3(512, PP), dim3(256), 0, stream>>>(
        node_attrs, PP * DD, W_node_props, prop, PP * DD, NN / 16, DD, DD * DD, DD);

    // -------- Graph reasoning steps --------
    nsm_dist_init<<<dim3((NN + 255) / 256), dim3(256), 0, stream>>>(npg, node_indices, dist);
    for (int t = 0; t < NSTEP; ++t) {
        nsm_psim       <<<dim3(BG), dim3(32), 0, stream>>>(instr, pemb, psim, t);
        nsm_node_score <<<dim3(NN), dim3(128), 0, stream>>>(prop, psim, instr, w_node_score,
                                                            node_indices, nscore, t);
        nsm_seg_softmax<<<dim3(BG), dim3(256), 0, stream>>>(nscore, pstates);
        nsm_zero       <<<dim3((NN * DD) / 256), dim3(256), 0, stream>>>(agg, (long)NN * DD);
        nsm_edge_scatter<<<dim3(EE / 2), dim3(256), 0, stream>>>(eproj, instr, dist,
                                                                 edge_src, edge_dst, edge_bidx, agg, t);
        nsm_rel_score  <<<dim3(NN), dim3(128), 0, stream>>>(agg, w_rel_score, rscore);
        nsm_seg_softmax<<<dim3(BG), dim3(256), 0, stream>>>(rscore, prel);
        nsm_dist_update<<<dim3((NN + 255) / 256), dim3(256), 0, stream>>>(psim, prel, pstates,
                                                                          node_indices, dist);
    }

    // -------- Final aggregation + linear head --------
    nsm_zero        <<<dim3((BG * DD + 255) / 256), dim3(256), 0, stream>>>(aggregated, (long)BG * DD);
    nsm_final_agg   <<<dim3(BG * 10), dim3(128), 0, stream>>>(node_attrs, psim, dist, aggregated);
    nsm_final_linear<<<dim3(BG), dim3(256), 0, stream>>>(encoded, aggregated, lin_W, lin_b, outp);
}
